// esm_model_773094113862
// MI455X (gfx1250) — matrix operand order repaired and checked by reading, not compiled, not run on hardware
//
#include <hip/hip_runtime.h>
#include <hip/hip_bf16.h>
#include <math.h>

typedef __attribute__((ext_vector_type(16))) _Float16 v16h;
typedef __attribute__((ext_vector_type(8)))  _Float16 v8h;
typedef __attribute__((ext_vector_type(8)))  float    v8f;

#define N_NODES  20000
#define N_EDGES  320000
#define N_GRAPHS 64
#define D_IN     1280
#define D_H      512
#define D_FC     1024
#define OUT_DIMS 5000
#define BN_EPS   1e-5f

// ---------------- utility kernels ----------------

__global__ void k_init_f(float* p, float v, int n) {
  int i = blockIdx.x * blockDim.x + threadIdx.x;
  if (i < n) p[i] = v;
}

__global__ void k_deg_accum(const int* __restrict__ dst, float* deg, int e) {
  int i = blockIdx.x * blockDim.x + threadIdx.x;
  if (i < e) atomicAdd(&deg[dst[i]], 1.0f);
}

__global__ void k_deg_to_dinv(float* deg, int n) {
  int i = blockIdx.x * blockDim.x + threadIdx.x;
  if (i < n) deg[i] = rsqrtf(deg[i]);
}

__global__ void k_edge_coef(const int* __restrict__ src, const int* __restrict__ dst,
                            const float* __restrict__ dinv, float* coef, int e) {
  int i = blockIdx.x * blockDim.x + threadIdx.x;
  if (i < e) coef[i] = dinv[src[i]] * dinv[dst[i]];
}

// hout[n][d] = hlin[n][d] * dinv[n]^2 + bias[d]   (self-loop + bias seed)
__global__ void k_selfloop_init(const float* __restrict__ hlin, const float* __restrict__ dinv,
                                const float* __restrict__ bias, float* __restrict__ hout,
                                int nodes, int dim) {
  int i = blockIdx.x * blockDim.x + threadIdx.x;
  int per = dim >> 2;
  if (i >= nodes * per) return;
  int n = i / per;
  int d4 = (i - n * per) << 2;
  float di = dinv[n];
  float s = di * di;
  float4 h = *(const float4*)(hlin + n * dim + d4);
  float4 o;
  o.x = h.x * s + bias[d4 + 0];
  o.y = h.y * s + bias[d4 + 1];
  o.z = h.z * s + bias[d4 + 2];
  o.w = h.w * s + bias[d4 + 3];
  *(float4*)(hout + n * dim + d4) = o;
}

// hout[dst[e]][:] += coef[e] * hlin[src[e]][:]   (4 dims per thread)
__global__ void k_edge_scatter(const int* __restrict__ src, const int* __restrict__ dst,
                               const float* __restrict__ coef, const float* __restrict__ hlin,
                               float* __restrict__ hout, int dim) {
  int i = blockIdx.x * blockDim.x + threadIdx.x;
  int per = dim >> 2;
  if (i >= N_EDGES * per) return;
  int e = i / per;
  int d4 = (i - e * per) << 2;
  int s = src[e], d = dst[e];
  float c = coef[e];
  float4 h = *(const float4*)(hlin + s * dim + d4);
  float* out = hout + d * dim + d4;
  atomicAdd(out + 0, h.x * c);
  atomicAdd(out + 1, h.y * c);
  atomicAdd(out + 2, h.z * c);
  atomicAdd(out + 3, h.w * c);
}

// g[batch[n]][d] = max(g, relu(h[n][d]))  via uint atomicMax (values >= 0)
__global__ void k_pool_max(const float* __restrict__ h, const int* __restrict__ batch,
                           float* __restrict__ g, int nodes, int dim) {
  int i = blockIdx.x * blockDim.x + threadIdx.x;
  if (i >= nodes * dim) return;
  int n = i / dim;
  int d = i - n * dim;
  float v = fmaxf(h[i], 0.0f);
  atomicMax((unsigned int*)&g[batch[n] * dim + d], __float_as_uint(v));
}

// BT[n][k] = (f16) B[k][n]   -- LDS-tiled transpose + fp32->f16 convert
__global__ __launch_bounds__(256) void k_transpose_f16(const float* __restrict__ B,
                                                       _Float16* __restrict__ BT,
                                                       int K, int N) {
  __shared__ _Float16 tile[32][33];
  int bk = blockIdx.x * 32;
  int bn = blockIdx.y * 32;
  int tx = threadIdx.x & 31;
  int ty = threadIdx.x >> 5;
#pragma unroll
  for (int r = 0; r < 32; r += 8) {
    int k = bk + ty + r;
    int n = bn + tx;
    float v = (k < K && n < N) ? B[(long long)k * N + n] : 0.0f;
    tile[tx][ty + r] = (_Float16)v;          // tile[n_local][k_local]
  }
  __syncthreads();
#pragma unroll
  for (int r = 0; r < 32; r += 8) {
    int n = bn + ty + r;
    int k = bk + tx;
    if (n < N && k < K) BT[(long long)n * K + k] = tile[ty + r][tx];
  }
}

// ---------------- WMMA GEMM ----------------
// C[M,N] = A[M,K] @ B[K,N], with B pre-transposed+converted: BT[N][K] (f16, ld=K).
// A fp32 (optionally relu'd on load), f16 WMMA with fp32 accumulation.
// Wave computes a 16(M) x 64(N) tile: 1 shared A fragment, 4 B fragments, 4 WMMAs/K-step.
// mode 0: store acc;  mode 1: relu(acc+bias);  mode 2: sigmoid((acc+bias)*gamma/sqrt(1+eps)+beta)
// Requires M % 16 == 0, K % 32 == 0.  N may be ragged (guarded).
__global__ __launch_bounds__(256) void k_wmma_gemm(
    const float* __restrict__ A, const _Float16* __restrict__ BT, float* __restrict__ C,
    int M, int N, int K, int mode, int relu_a,
    const float* __restrict__ bias, const float* __restrict__ gamma,
    const float* __restrict__ beta) {
  int wave = threadIdx.x >> 5;
  int lane = threadIdx.x & 31;
  int tilesN = (N + 63) >> 6;                 // 64-wide N groups
  int tilesM = M >> 4;
  int t = blockIdx.x * 8 + wave;
  if (t >= tilesM * tilesN) return;           // wave-uniform exit; EXEC stays all-1
  int tileM = t / tilesN;
  int tN = t - tileM * tilesN;

  int halfsel = lane >> 4;                    // 0 or 1
  int row  = (tileM << 4) + (lane & 15);      // A row held by this lane
  int col0 = (tN << 6) + (lane & 15);         // first of 4 B/C columns for this lane
  int kA = halfsel * 8;                       // A lane K base (ISA 16-bit A layout)
  int kB = halfsel * 8;                       // B lane K base (same layout as A)

  // clamped column indices for loads (garbage stays in guarded lanes)
  int cc[4];
#pragma unroll
  for (int i = 0; i < 4; ++i) {
    int c = col0 + 16 * i;
    cc[i] = c < N ? c : N - 1;
  }

  v8f acc[4] = {};
  for (int k0 = 0; k0 < K; k0 += 32) {
    const float* Ap = A + (long long)row * K + k0 + kA;
    float4 a0 = *(const float4*)(Ap + 0);
    float4 a1 = *(const float4*)(Ap + 4);
    float4 a2 = *(const float4*)(Ap + 16);
    float4 a3 = *(const float4*)(Ap + 20);
    if (relu_a) {
      a0.x = fmaxf(a0.x, 0.f); a0.y = fmaxf(a0.y, 0.f); a0.z = fmaxf(a0.z, 0.f); a0.w = fmaxf(a0.w, 0.f);
      a1.x = fmaxf(a1.x, 0.f); a1.y = fmaxf(a1.y, 0.f); a1.z = fmaxf(a1.z, 0.f); a1.w = fmaxf(a1.w, 0.f);
      a2.x = fmaxf(a2.x, 0.f); a2.y = fmaxf(a2.y, 0.f); a2.z = fmaxf(a2.z, 0.f); a2.w = fmaxf(a2.w, 0.f);
      a3.x = fmaxf(a3.x, 0.f); a3.y = fmaxf(a3.y, 0.f); a3.z = fmaxf(a3.z, 0.f); a3.w = fmaxf(a3.w, 0.f);
    }
    v16h a;
    a[0]  = (_Float16)a0.x; a[1]  = (_Float16)a0.y; a[2]  = (_Float16)a0.z; a[3]  = (_Float16)a0.w;
    a[4]  = (_Float16)a1.x; a[5]  = (_Float16)a1.y; a[6]  = (_Float16)a1.z; a[7]  = (_Float16)a1.w;
    a[8]  = (_Float16)a2.x; a[9]  = (_Float16)a2.y; a[10] = (_Float16)a2.z; a[11] = (_Float16)a2.w;
    a[12] = (_Float16)a3.x; a[13] = (_Float16)a3.y; a[14] = (_Float16)a3.z; a[15] = (_Float16)a3.w;

#pragma unroll
    for (int i = 0; i < 4; ++i) {
      const _Float16* Bp = BT + (long long)cc[i] * K + k0 + kB;
      v8h blo = *(const v8h*)(Bp + 0);
      v8h bhi = *(const v8h*)(Bp + 16);
      v16h b = __builtin_shufflevector(blo, bhi, 0, 1, 2, 3, 4, 5, 6, 7,
                                       8, 9, 10, 11, 12, 13, 14, 15);
      acc[i] = __builtin_amdgcn_wmma_f32_16x16x32_f16(
          /*neg_a=*/false, a, /*neg_b=*/false, b,
          /*c_mod=*/(short)0, acc[i], /*reuse_a=*/false, /*reuse_b=*/false);
    }
  }

#pragma unroll
  for (int i = 0; i < 4; ++i) {
    int col = col0 + 16 * i;
    if (col >= N) continue;
    float bi = 0.f, scale = 0.f, shift = 0.f;
    if (mode >= 1) bi = bias[col];
    if (mode == 2) { scale = gamma[col] * rsqrtf(1.0f + BN_EPS); shift = beta[col]; }
#pragma unroll
    for (int r8 = 0; r8 < 8; ++r8) {
      int r = (tileM << 4) + r8 + (halfsel << 3);
      float v = acc[i][r8];
      if (mode == 1) v = fmaxf(v + bi, 0.f);
      else if (mode == 2) {
        v = (v + bi) * scale + shift;
        v = 1.0f / (1.0f + expf(-v));
      }
      C[(long long)r * N + col] = v;
    }
  }
}

// ---------------- launch ----------------

extern "C" void kernel_launch(void* const* d_in, const int* in_sizes, int n_in,
                              void* d_out, int out_size, void* d_ws, size_t ws_size,
                              hipStream_t stream) {
  const float* x     = (const float*)d_in[0];
  const int*   edge  = (const int*)d_in[1];
  const int*   batch = (const int*)d_in[2];
  const float* Wg1   = (const float*)d_in[3];
  const float* bg1   = (const float*)d_in[4];
  const float* Wg2   = (const float*)d_in[5];
  const float* bg2   = (const float*)d_in[6];
  const float* W1    = (const float*)d_in[7];
  const float* b1    = (const float*)d_in[8];
  const float* W2    = (const float*)d_in[9];
  const float* b2    = (const float*)d_in[10];
  const float* gamma = (const float*)d_in[11];
  const float* beta  = (const float*)d_in[12];
  float* out = (float*)d_out;

  char* ws = (char*)d_ws;
  size_t off = 0;
  auto take = [&](size_t bytes) -> char* {
    char* p = ws + off;
    off = (off + bytes + 255) & ~(size_t)255;
    return p;
  };
  float*     HLIN = (float*)take((size_t)N_NODES * D_H * 4);      // linear xform output
  float*     HOUT = (float*)take((size_t)N_NODES * D_H * 4);      // aggregated output
  float*     DINV = (float*)take((size_t)N_NODES * 4);            // degree -> rsqrt(degree)
  float*     COEF = (float*)take((size_t)N_EDGES * 4);
  float*     G    = (float*)take((size_t)N_GRAPHS * D_H * 4);
  float*     H1   = (float*)take((size_t)N_GRAPHS * D_FC * 4);
  _Float16*  WT   = (_Float16*)take((size_t)D_FC * OUT_DIMS * 2); // transposed f16 weight (reused)

  const int* src = edge;
  const int* dst = edge + N_EDGES;

  const int T = 256;
  auto cdiv = [](int a, int b) { return (a + b - 1) / b; };

  // degree / normalization
  k_init_f<<<cdiv(N_NODES, T), T, 0, stream>>>(DINV, 1.0f, N_NODES);
  k_deg_accum<<<cdiv(N_EDGES, T), T, 0, stream>>>(dst, DINV, N_EDGES);
  k_deg_to_dinv<<<cdiv(N_NODES, T), T, 0, stream>>>(DINV, N_NODES);
  k_edge_coef<<<cdiv(N_EDGES, T), T, 0, stream>>>(src, dst, DINV, COEF, N_EDGES);

  // ---- layer 1 ----
  k_transpose_f16<<<dim3(D_IN / 32, D_H / 32), T, 0, stream>>>(Wg1, WT, D_IN, D_H);
  {
    int tiles = (N_NODES / 16) * (D_H / 64);
    k_wmma_gemm<<<cdiv(tiles, 8), T, 0, stream>>>(x, WT, HLIN, N_NODES, D_H, D_IN,
                                                  0, 0, nullptr, nullptr, nullptr);
  }
  k_selfloop_init<<<cdiv(N_NODES * (D_H / 4), T), T, 0, stream>>>(HLIN, DINV, bg1, HOUT, N_NODES, D_H);
  k_edge_scatter<<<cdiv(N_EDGES * (D_H / 4), T), T, 0, stream>>>(src, dst, COEF, HLIN, HOUT, D_H);

  // ---- layer 2 (A = relu(HOUT) applied on load) ----
  k_transpose_f16<<<dim3(D_H / 32, D_H / 32), T, 0, stream>>>(Wg2, WT, D_H, D_H);
  {
    int tiles = (N_NODES / 16) * (D_H / 64);
    k_wmma_gemm<<<cdiv(tiles, 8), T, 0, stream>>>(HOUT, WT, HLIN, N_NODES, D_H, D_H,
                                                  0, 1, nullptr, nullptr, nullptr);
  }
  k_selfloop_init<<<cdiv(N_NODES * (D_H / 4), T), T, 0, stream>>>(HLIN, DINV, bg2, HOUT, N_NODES, D_H);
  k_edge_scatter<<<cdiv(N_EDGES * (D_H / 4), T), T, 0, stream>>>(src, dst, COEF, HLIN, HOUT, D_H);

  // ---- global max pool (relu fused) ----
  k_init_f<<<cdiv(N_GRAPHS * D_H, T), T, 0, stream>>>(G, 0.0f, N_GRAPHS * D_H);
  k_pool_max<<<cdiv(N_NODES * D_H, T), T, 0, stream>>>(HOUT, batch, G, N_NODES, D_H);

  // ---- FC head ----
  k_transpose_f16<<<dim3(D_H / 32, D_FC / 32), T, 0, stream>>>(W1, WT, D_H, D_FC);
  {
    int tiles = (N_GRAPHS / 16) * (D_FC / 64);
    k_wmma_gemm<<<cdiv(tiles, 8), T, 0, stream>>>(G, WT, H1, N_GRAPHS, D_FC, D_H,
                                                  1, 0, b1, nullptr, nullptr);
  }
  k_transpose_f16<<<dim3(D_FC / 32, (OUT_DIMS + 31) / 32), T, 0, stream>>>(W2, WT, D_FC, OUT_DIMS);
  {
    int tiles = (N_GRAPHS / 16) * ((OUT_DIMS + 63) / 64);
    k_wmma_gemm<<<cdiv(tiles, 8), T, 0, stream>>>(H1, WT, out, N_GRAPHS, OUT_DIMS, D_FC,
                                                  2, 0, b2, gamma, beta);
  }
  (void)in_sizes; (void)n_in; (void)out_size; (void)ws_size;
}
